// SVDHead_88802743812377
// MI455X (gfx1250) — matrix operand order repaired and checked by reading, not compiled, not run on hardware
//
#include <hip/hip_runtime.h>

// ---------------- constants (match reference shapes) ----------------
#define B_   2
#define N_   1024
#define M_   1024
#define C_   512
#define K_   16
#define K1_  6
#define NKP  256
#define DIM_ 256   // conv1 out channels
#define D2_  512   // conv2 out channels

typedef unsigned short u16;
typedef __bf16 v16bf __attribute__((ext_vector_type(16)));
typedef float  v8f   __attribute__((ext_vector_type(8)));

union ABfrag { v16bf v; unsigned u[8]; };
union CDfrag { v8f   v; float   f[8]; };

__device__ __forceinline__ u16 f2bf(float f){
  unsigned u = __float_as_uint(f);
  unsigned r = u + 0x7FFFu + ((u >> 16) & 1u);   // round-to-nearest-even
  return (u16)(r >> 16);
}
__device__ __forceinline__ float bf2f(u16 h){ return __uint_as_float(((unsigned)h) << 16); }
__device__ __forceinline__ float lrelu(float v){ return v > 0.f ? v : 0.2f * v; }

// ---------------- embedding transpose to bf16: out[b][n][c] ----------------
__global__ void k_embT(const float* __restrict__ emb, u16* __restrict__ outT){
  int idx = blockIdx.x * blockDim.x + threadIdx.x;           // total B_*N_*C_
  int c = idx & (C_ - 1);
  int n = (idx >> 9) & (N_ - 1);
  int b = idx >> 19;
  outT[idx] = f2bf(emb[((size_t)(b * C_ + c)) * N_ + n]);
}

// ---------------- xx[b,n] = sum_c emb[b,c,n]^2 ----------------
__global__ void k_sqsum(const float* __restrict__ emb, float* __restrict__ out){
  int idx = blockIdx.x * blockDim.x + threadIdx.x;           // total B_*N_
  int n = idx & (N_ - 1);
  int b = idx >> 10;
  float s = 0.f;
  for (int c = 0; c < C_; c++){ float v = emb[((size_t)(b * C_ + c)) * N_ + n]; s += v * v; }
  out[idx] = s;
}

// ---------------- pairwise distance via WMMA bf16, 32x32 tile per wave ----------------
// dmap[b,n,m] = masked_sqrt(xx[n] - 2*dot + yy[m])
__global__ __launch_bounds__(32) void k_pair_wmma(const u16* __restrict__ srcT,
                                                  const u16* __restrict__ tgtT,
                                                  const float* __restrict__ xx,
                                                  const float* __restrict__ yy,
                                                  float* __restrict__ dmap){
  int b  = blockIdx.z;
  int n0 = blockIdx.y << 5;
  int m0 = blockIdx.x << 5;
  int l  = threadIdx.x;
  int hi = (l >= 16) ? 1 : 0;
  int lm = l & 15;
  const unsigned* Ap0 = (const unsigned*)(srcT + ((size_t)(b * N_ + n0      + lm)) * C_);
  const unsigned* Ap1 = (const unsigned*)(srcT + ((size_t)(b * N_ + n0 + 16 + lm)) * C_);
  const unsigned* Bp0 = (const unsigned*)(tgtT + ((size_t)(b * M_ + m0      + lm)) * C_);
  const unsigned* Bp1 = (const unsigned*)(tgtT + ((size_t)(b * M_ + m0 + 16 + lm)) * C_);
  CDfrag a00, a01, a10, a11;
  #pragma unroll
  for (int r = 0; r < 8; r++){ a00.f[r] = 0.f; a01.f[r] = 0.f; a10.f[r] = 0.f; a11.f[r] = 0.f; }
  for (int c0 = 0; c0 < C_; c0 += 32){
    ABfrag A0, A1, Bb0, Bb1;
    #pragma unroll
    for (int v = 0; v < 8; v++){
      int klo = (v < 4) ? ((hi ? 8 : 0) + 2 * v) : ((hi ? 24 : 16) + 2 * (v - 4));
      A0.u[v]  = Ap0[(c0 + klo) >> 1];
      A1.u[v]  = Ap1[(c0 + klo) >> 1];
      int kb   = klo;
      Bb0.u[v] = Bp0[(c0 + kb) >> 1];
      Bb1.u[v] = Bp1[(c0 + kb) >> 1];
    }
    __builtin_prefetch(Ap0 + ((c0 + 32) >> 1), 0, 1);
    __builtin_prefetch(Ap1 + ((c0 + 32) >> 1), 0, 1);
    __builtin_prefetch(Bp0 + ((c0 + 32) >> 1), 0, 1);
    __builtin_prefetch(Bp1 + ((c0 + 32) >> 1), 0, 1);
    a00.v = __builtin_amdgcn_wmma_f32_16x16x32_bf16(false, A0.v, false, Bb0.v, (short)0, a00.v, false, false);
    a01.v = __builtin_amdgcn_wmma_f32_16x16x32_bf16(false, A0.v, false, Bb1.v, (short)0, a01.v, false, false);
    a10.v = __builtin_amdgcn_wmma_f32_16x16x32_bf16(false, A1.v, false, Bb0.v, (short)0, a10.v, false, false);
    a11.v = __builtin_amdgcn_wmma_f32_16x16x32_bf16(false, A1.v, false, Bb1.v, (short)0, a11.v, false, false);
  }
  auto emit = [&](CDfrag& acc, int nb, int mb){
    #pragma unroll
    for (int r = 0; r < 8; r++){
      int n = nb + r + (hi ? 8 : 0);
      int m = mb + lm;
      float d   = xx[b * N_ + n] - 2.f * acc.f[r] + yy[b * M_ + m];
      float dsq = d > 0.f ? d : 0.f;
      dmap[((size_t)(b * N_ + n)) * M_ + m] = (dsq <= 0.f) ? 0.f : sqrtf(dsq);
    }
  };
  emit(a00, n0, m0);
  emit(a01, n0, m0 + 16);
  emit(a10, n0 + 16, m0);
  emit(a11, n0 + 16, m0 + 16);
}

// ---------------- row softmax of (-x), optional argmax of result ----------------
__global__ __launch_bounds__(256) void k_softmax_neg(const float* __restrict__ in,
                                                     float* __restrict__ out,
                                                     int* __restrict__ amax){
  int row = blockIdx.x;                                      // B_*N_ rows of M_
  int t = threadIdx.x;
  const float* x = in + (size_t)row * M_;
  float* o = out + (size_t)row * M_;
  __shared__ float shv[256];
  __shared__ int   shi[256];
  float best = -3.0e38f; int bidx = 0x7fffffff;
  for (int m = t; m < M_; m += 256){
    float v = -x[m];
    if (v > best || (v == best && m < bidx)){ best = v; bidx = m; }
  }
  shv[t] = best; shi[t] = bidx; __syncthreads();
  for (int st = 128; st > 0; st >>= 1){
    if (t < st){
      float v2 = shv[t + st]; int i2 = shi[t + st];
      if (v2 > shv[t] || (v2 == shv[t] && i2 < shi[t])){ shv[t] = v2; shi[t] = i2; }
    }
    __syncthreads();
  }
  float mx = shv[0]; int mi = shi[0]; __syncthreads();
  float s = 0.f;
  for (int m = t; m < M_; m += 256) s += expf(-x[m] - mx);
  shv[t] = s; __syncthreads();
  for (int st = 128; st > 0; st >>= 1){ if (t < st) shv[t] += shv[t + st]; __syncthreads(); }
  float inv = 1.f / shv[0];
  for (int m = t; m < M_; m += 256) o[m] = expf(-x[m] - mx) * inv;
  if (amax && t == 0) amax[row] = mi;
}

// ---------------- refined map: 25-gather per element ----------------
__global__ void k_refine(const float* __restrict__ scores, const float* __restrict__ dmap,
                         const int* __restrict__ sidx1, const int* __restrict__ idx2,
                         float* __restrict__ refined){
  int idx = blockIdx.x * blockDim.x + threadIdx.x;           // total B_*N_*M_
  int mm = idx & (M_ - 1);
  int n  = (idx >> 10) & (N_ - 1);
  int b  = idx >> 20;
  int rbase = (b * N_ + n) * K1_;
  int cbase = (b * M_ + mm) * K1_;
  int r[5], cc[5];
  #pragma unroll
  for (int j = 0; j < 5; j++){ r[j] = sidx1[rbase + 1 + j]; cc[j] = idx2[cbase + 1 + j]; }
  float s = 0.f;
  #pragma unroll
  for (int j = 0; j < 5; j++){
    const float* rp = scores + (size_t)r[j] * M_;
    #pragma unroll
    for (int jj = 0; jj < 5; jj++) s += rp[cc[jj]];
  }
  float sk = s * 0.2f;                                       // / (K1-1)
  refined[idx] = expf(0.7f - sk) * dmap[idx];
}

// ---------------- src_corr[b,c,n] = sum_m tgt[b,c,m]*rmm[b,n,m] ----------------
__global__ __launch_bounds__(128) void k_src_corr(const float* __restrict__ tgt,
                                                  const float* __restrict__ rmm,
                                                  float* __restrict__ corr){
  int row = blockIdx.x;                                      // b*N_+n
  int b = row >> 10, n = row & (N_ - 1);
  int t = threadIdx.x;
  const float* rp = rmm + (size_t)row * M_;
  float a0 = 0.f, a1 = 0.f, a2 = 0.f;
  for (int m = t; m < M_; m += 128){
    float rv = rp[m];
    a0 += tgt[(b * 3 + 0) * M_ + m] * rv;
    a1 += tgt[(b * 3 + 1) * M_ + m] * rv;
    a2 += tgt[(b * 3 + 2) * M_ + m] * rv;
  }
  __shared__ float sh[3][128];
  sh[0][t] = a0; sh[1][t] = a1; sh[2][t] = a2; __syncthreads();
  for (int st = 64; st > 0; st >>= 1){
    if (t < st){ sh[0][t] += sh[0][t+st]; sh[1][t] += sh[1][t+st]; sh[2][t] += sh[2][t+st]; }
    __syncthreads();
  }
  if (t < 3) corr[(b * 3 + t) * N_ + n] = sh[t][0];
}

// ---------------- knn distance tensors (B,N,K,3) ----------------
__global__ void k_knn_dist(const float* __restrict__ src, const float* __restrict__ srcCorr,
                           const float* __restrict__ src_knn, const int* __restrict__ src_idx,
                           float* __restrict__ knnD, float* __restrict__ sknnD){
  int idx = blockIdx.x * blockDim.x + threadIdx.x;           // total B_*N_*K_
  int n = (idx >> 4) & (N_ - 1);
  int b = idx >> 14;
  int gi = src_idx[idx];
  int b2 = gi >> 10, n2 = gi & (N_ - 1);
  #pragma unroll
  for (int c = 0; c < 3; c++){
    knnD [idx * 3 + c] = srcCorr[(b * 3 + c) * N_ + n] - srcCorr[(b2 * 3 + c) * N_ + n2];
    sknnD[idx * 3 + c] = src    [(b * 3 + c) * N_ + n] - src_knn[idx * 3 + c];
  }
}

// ---------------- conv1: 3ch -> 256ch, 3-tap over K dim; z1 bf16 [b][i][n][o] ----------------
__global__ __launch_bounds__(256) void k_conv1(const float* __restrict__ X,
                                               const float* __restrict__ w1,
                                               const float* __restrict__ b1,
                                               u16* __restrict__ z1){
  int gid = blockIdx.x;                                      // (b*16+i)*N_+n, o = thread
  int n = gid & (N_ - 1);
  int i = (gid >> 10) & 15;
  int b = gid >> 14;
  __shared__ float xv[9];
  int t = threadIdx.x;
  if (t < 9){
    int dh = t / 3, ci = t % 3;
    int iy = i + dh - 1;
    xv[t] = (iy >= 0 && iy < K_) ? X[((size_t)((b * N_ + n) * K_ + iy)) * 3 + ci] : 0.f;
  }
  __syncthreads();
  int o = t;
  float acc = b1[o];
  #pragma unroll
  for (int ci = 0; ci < 3; ci++)
    #pragma unroll
    for (int dh = 0; dh < 3; dh++)
      acc += xv[dh * 3 + ci] * w1[(o * 3 + ci) * 3 + dh];
  z1[(size_t)gid * DIM_ + o] = f2bf(acc);
}

// ---------------- w2 repack to bf16 [o][dh][ci] ----------------
__global__ void k_w2conv(const float* __restrict__ w2, u16* __restrict__ w2b){
  int idx = blockIdx.x * blockDim.x + threadIdx.x;           // total 512*3*256
  int ci = idx & 255;
  int dh = (idx >> 8) % 3;
  int o  = idx / 768;
  w2b[idx] = f2bf(w2[(o * DIM_ + ci) * 3 + dh]);
}

// ---------------- conv2 via WMMA: 32x32 (o,n) tile per wave, K=768 reduction ----------------
__global__ __launch_bounds__(32) void k_conv2_wmma(const u16* __restrict__ h1,
                                                   const u16* __restrict__ w2b,
                                                   const float* __restrict__ b2,
                                                   u16* __restrict__ z2){
  int bi = blockIdx.z; int b = bi >> 4; int i = bi & 15;
  int o0 = blockIdx.y << 5;
  int n0 = blockIdx.x << 5;
  int l = threadIdx.x;
  int hi = (l >= 16) ? 1 : 0;
  int lm = l & 15;
  const unsigned* Aw0 = (const unsigned*)(w2b + (size_t)(o0      + lm) * 768);
  const unsigned* Aw1 = (const unsigned*)(w2b + (size_t)(o0 + 16 + lm) * 768);
  CDfrag a00, a01, a10, a11;
  #pragma unroll
  for (int r = 0; r < 8; r++){ a00.f[r] = 0.f; a01.f[r] = 0.f; a10.f[r] = 0.f; a11.f[r] = 0.f; }
  for (int ks = 0; ks < 24; ks++){
    int k0  = ks << 5;
    int dh  = k0 >> 8;          // 32 | 256 so each step is inside one tap
    int ci0 = k0 & 255;
    int iy  = i + dh - 1;
    bool valid = (iy >= 0 && iy < K_);
    int iyc = valid ? iy : 0;
    const unsigned* Bx0 = (const unsigned*)(h1 + ((size_t)((b * K_ + iyc) * N_ + n0      + lm)) * DIM_);
    const unsigned* Bx1 = (const unsigned*)(h1 + ((size_t)((b * K_ + iyc) * N_ + n0 + 16 + lm)) * DIM_);
    ABfrag A0, A1, Bb0, Bb1;
    #pragma unroll
    for (int v = 0; v < 8; v++){
      int klo = (v < 4) ? ((hi ? 8 : 0) + 2 * v) : ((hi ? 24 : 16) + 2 * (v - 4));
      A0.u[v]  = Aw0[(k0 + klo) >> 1];
      A1.u[v]  = Aw1[(k0 + klo) >> 1];
      int kb   = klo;
      Bb0.u[v] = valid ? Bx0[(ci0 + kb) >> 1] : 0u;
      Bb1.u[v] = valid ? Bx1[(ci0 + kb) >> 1] : 0u;
    }
    __builtin_prefetch(Aw0 + ((k0 + 32) >> 1), 0, 1);
    __builtin_prefetch(Aw1 + ((k0 + 32) >> 1), 0, 1);
    a00.v = __builtin_amdgcn_wmma_f32_16x16x32_bf16(false, A0.v, false, Bb0.v, (short)0, a00.v, false, false);
    a01.v = __builtin_amdgcn_wmma_f32_16x16x32_bf16(false, A0.v, false, Bb1.v, (short)0, a01.v, false, false);
    a10.v = __builtin_amdgcn_wmma_f32_16x16x32_bf16(false, A1.v, false, Bb0.v, (short)0, a10.v, false, false);
    a11.v = __builtin_amdgcn_wmma_f32_16x16x32_bf16(false, A1.v, false, Bb1.v, (short)0, a11.v, false, false);
  }
  auto emit = [&](CDfrag& acc, int ob, int nb){
    #pragma unroll
    for (int r = 0; r < 8; r++){
      int o = ob + r + (hi ? 8 : 0);
      int n = nb + lm;
      z2[((size_t)((b * K_ + i) * N_ + n)) * D2_ + o] = f2bf(acc.f[r] + b2[o]);
    }
  };
  emit(a00, o0, n0);
  emit(a01, o0, n0 + 16);
  emit(a10, o0 + 16, n0);
  emit(a11, o0 + 16, n0 + 16);
}

// ---------------- per-channel stats, layout [row][ch] bf16 ----------------
__global__ __launch_bounds__(256) void k_stats_last(const u16* __restrict__ x, int nchan, int rows,
                                                    float* __restrict__ mu, float* __restrict__ var){
  int ch = blockIdx.x, t = threadIdx.x;
  float s = 0.f, s2 = 0.f;
  for (int r = t; r < rows; r += 256){
    float v = bf2f(x[(size_t)r * nchan + ch]);
    s += v; s2 += v * v;
  }
  __shared__ float sh[256], sh2[256];
  sh[t] = s; sh2[t] = s2; __syncthreads();
  for (int st = 128; st > 0; st >>= 1){
    if (t < st){ sh[t] += sh[t+st]; sh2[t] += sh2[t+st]; } __syncthreads();
  }
  if (t == 0){ float m = sh[0] / rows; mu[ch] = m; var[ch] = sh2[0] / rows - m * m; }
}

// ---------------- in-place BN + lrelu, layout [row][ch] bf16 ----------------
__global__ void k_bn_lrelu_last(u16* __restrict__ x, const float* __restrict__ g,
                                const float* __restrict__ be, const float* __restrict__ mu,
                                const float* __restrict__ var, int nchan){
  size_t idx = (size_t)blockIdx.x * blockDim.x + threadIdx.x;
  int ch = (int)(idx % nchan);
  float v = bf2f(x[idx]);
  v = g[ch] * (v - mu[ch]) * rsqrtf(var[ch] + 1e-5f) + be[ch];
  x[idx] = f2bf(lrelu(v));
}

// ---------------- x2 = h2x - lrelu(bn(z2y)) ----------------
__global__ void k_bn_lrelu_sub(u16* __restrict__ A, const u16* __restrict__ Bz,
                               const float* __restrict__ g, const float* __restrict__ be,
                               const float* __restrict__ mu, const float* __restrict__ var){
  size_t idx = (size_t)blockIdx.x * blockDim.x + threadIdx.x;
  int ch = (int)(idx % D2_);
  float v = bf2f(Bz[idx]);
  v = lrelu(g[ch] * (v - mu[ch]) * rsqrtf(var[ch] + 1e-5f) + be[ch]);
  A[idx] = f2bf(bf2f(A[idx]) - v);
}

// ---------------- 1x1 conv 512->16, input bf16 [b][i][n][ci], out f32 (B,16,N,16) ----------------
__global__ void k_conv1x1_bf(const u16* __restrict__ x2, const float* __restrict__ w,
                             const float* __restrict__ bias, float* __restrict__ out){
  int idx = blockIdx.x * blockDim.x + threadIdx.x;           // total B_*16*N_*16
  int i = idx & 15;
  int n = (idx >> 4) & (N_ - 1);
  int o = (idx >> 14) & 15;
  int b = idx >> 18;
  const u16* xp = x2 + ((size_t)((b * K_ + i) * N_ + n)) * D2_;
  const float* wp = w + o * D2_;
  float acc = bias[o];
  for (int ci = 0; ci < D2_; ci++) acc += bf2f(xp[ci]) * wp[ci];
  out[((size_t)((b * 16 + o) * N_ + n)) * 16 + i] = acc;
}

// ---------------- 1x1 conv 16->16, f32 (B,16,N,16) -> (B,16,N,16) ----------------
__global__ void k_conv1x1_f32(const float* __restrict__ in, const float* __restrict__ w,
                              const float* __restrict__ bias, float* __restrict__ out){
  int idx = blockIdx.x * blockDim.x + threadIdx.x;
  int i = idx & 15;
  int n = (idx >> 4) & (N_ - 1);
  int o = (idx >> 14) & 15;
  int b = idx >> 18;
  float acc = bias[o];
  #pragma unroll
  for (int ci = 0; ci < 16; ci++)
    acc += w[o * 16 + ci] * in[((size_t)((b * 16 + ci) * N_ + n)) * 16 + i];
  out[((size_t)((b * 16 + o) * N_ + n)) * 16 + i] = acc;
}

// ---------------- channel stats, layout [(b*O+o)*HW + p] f32 ----------------
__global__ __launch_bounds__(256) void k_stats_chw(const float* __restrict__ x, int O, int HW,
                                                   float* __restrict__ mu, float* __restrict__ var){
  int o = blockIdx.x, t = threadIdx.x;
  float s = 0.f, s2 = 0.f;
  for (int b = 0; b < B_; b++){
    const float* xp = x + (size_t)(b * O + o) * HW;
    for (int q = t; q < HW; q += 256){ float v = xp[q]; s += v; s2 += v * v; }
  }
  __shared__ float sh[256], sh2[256];
  sh[t] = s; sh2[t] = s2; __syncthreads();
  for (int st = 128; st > 0; st >>= 1){
    if (t < st){ sh[t] += sh[t+st]; sh2[t] += sh2[t+st]; } __syncthreads();
  }
  int cnt = B_ * HW;
  if (t == 0){ float m = sh[0] / cnt; mu[o] = m; var[o] = sh2[0] / cnt - m * m; }
}

__global__ void k_bn_lrelu_chw(float* __restrict__ x, const float* __restrict__ g,
                               const float* __restrict__ be, const float* __restrict__ mu,
                               const float* __restrict__ var, int O, int HW){
  size_t idx = (size_t)blockIdx.x * blockDim.x + threadIdx.x;
  int o = (int)((idx / HW) % O);
  float v = x[idx];
  x[idx] = lrelu(g[o] * (v - mu[o]) * rsqrtf(var[o] + 1e-5f) + be[o]);
}

// ---------------- softmax over last dim (16) of (B,16,N,16) ----------------
__global__ void k_softmax16(float* __restrict__ x){
  int idx = blockIdx.x * blockDim.x + threadIdx.x;           // total B_*16*N_
  float* p = x + (size_t)idx * 16;
  float mx = p[0];
  #pragma unroll
  for (int i = 1; i < 16; i++) mx = fmaxf(mx, p[i]);
  float s = 0.f;
  #pragma unroll
  for (int i = 0; i < 16; i++) s += expf(p[i] - mx);
  float inv = 1.f / s;
  #pragma unroll
  for (int i = 0; i < 16; i++) p[i] = expf(p[i] - mx) * inv;
}

// ---------------- s[b,o,n] = sum_i h*wt ----------------
__global__ void k_wsum(const float* __restrict__ h, const float* __restrict__ wt,
                       float* __restrict__ s){
  int idx = blockIdx.x * blockDim.x + threadIdx.x;           // total B_*16*N_
  const float* hp = h + (size_t)idx * 16;
  const float* wp = wt + (size_t)idx * 16;
  float acc = 0.f;
  #pragma unroll
  for (int i = 0; i < 16; i++) acc += hp[i] * wp[i];
  s[idx] = acc;
}

// ---------------- conv1d Cin->O over (B,Cin,N) ----------------
__global__ void k_conv1d(const float* __restrict__ in, const float* __restrict__ w,
                         const float* __restrict__ bias, float* __restrict__ out,
                         int Cin, int O){
  int idx = blockIdx.x * blockDim.x + threadIdx.x;           // total B_*O*N_
  int n = idx & (N_ - 1);
  int o = (idx >> 10) % O;
  int b = idx / (N_ * O);
  float acc = bias[o];
  for (int ci = 0; ci < Cin; ci++) acc += w[o * Cin + ci] * in[(size_t)(b * Cin + ci) * N_ + n];
  out[(size_t)(b * O + o) * N_ + n] = acc;
}

// ---------------- final: weight = 1 - tanh(|conv1d_8to1|) ----------------
__global__ void k_final_weight(const float* __restrict__ s2, const float* __restrict__ w,
                               const float* __restrict__ bias, float* __restrict__ wgt){
  int idx = blockIdx.x * blockDim.x + threadIdx.x;           // total B_*N_
  int n = idx & (N_ - 1);
  int b = idx >> 10;
  float acc = bias[0];
  #pragma unroll
  for (int ci = 0; ci < 8; ci++) acc += w[ci] * s2[(size_t)(b * 8 + ci) * N_ + n];
  wgt[idx] = 1.f - tanhf(fabsf(acc));
}

// ---------------- weighted rigid transform + 3x3 Jacobi SVD ----------------
__global__ __launch_bounds__(256) void k_rigid(const float* __restrict__ src,
                                               const float* __restrict__ corr,
                                               const float* __restrict__ wgt,
                                               float* __restrict__ outR, float* __restrict__ outT){
  int b = blockIdx.x, t = threadIdx.x;
  __shared__ float sh[256];
  __shared__ float acc[16];
  auto red = [&](float v, int slot){
    sh[t] = v; __syncthreads();
    for (int st = 128; st > 0; st >>= 1){ if (t < st) sh[t] += sh[t+st]; __syncthreads(); }
    if (t == 0) acc[slot] = sh[0];
    __syncthreads();
  };
  float sw = 0.f, sx[3] = {0,0,0}, sy[3] = {0,0,0};
  for (int n = t; n < N_; n += 256){
    float w = wgt[b * N_ + n];
    sw += w;
    #pragma unroll
    for (int c = 0; c < 3; c++){
      sx[c] += w * src [(b * 3 + c) * N_ + n];
      sy[c] += w * corr[(b * 3 + c) * N_ + n];
    }
  }
  red(sw, 0);
  red(sx[0], 1); red(sx[1], 2); red(sx[2], 3);
  red(sy[0], 4); red(sy[1], 5); red(sy[2], 6);
  __shared__ float cs[3], ct[3];
  if (t == 0){
    float inv = 1.f / acc[0];
    for (int c = 0; c < 3; c++){ cs[c] = acc[1+c] * inv; ct[c] = acc[4+c] * inv; }
  }
  __syncthreads();
  float h[9] = {0,0,0,0,0,0,0,0,0};
  for (int n = t; n < N_; n += 256){
    float w = wgt[b * N_ + n];
    float dx[3], dy[3];
    #pragma unroll
    for (int c = 0; c < 3; c++){
      dx[c] = (src [(b * 3 + c) * N_ + n] - cs[c]) * w;
      dy[c] =  corr[(b * 3 + c) * N_ + n] - ct[c];
    }
    #pragma unroll
    for (int i2 = 0; i2 < 3; i2++)
      #pragma unroll
      for (int j2 = 0; j2 < 3; j2++) h[i2*3+j2] += dx[i2] * dy[j2];
  }
  for (int q = 0; q < 9; q++) red(h[q], 7 + q);
  if (t == 0){
    float H[3][3];
    for (int i2 = 0; i2 < 3; i2++) for (int j2 = 0; j2 < 3; j2++) H[i2][j2] = acc[7 + i2*3 + j2];
    // S = H^T H
    float S[3][3];
    for (int i2 = 0; i2 < 3; i2++)
      for (int j2 = 0; j2 < 3; j2++){
        float s2v = 0.f;
        for (int k2 = 0; k2 < 3; k2++) s2v += H[k2][i2] * H[k2][j2];
        S[i2][j2] = s2v;
      }
    float V[3][3] = {{1,0,0},{0,1,0},{0,0,1}};
    const int pp[3] = {0,0,1}, qq[3] = {1,2,2};
    for (int sweep = 0; sweep < 30; sweep++){
      for (int pi = 0; pi < 3; pi++){
        int p = pp[pi], q = qq[pi];
        float apq = S[p][q];
        if (fabsf(apq) > 1e-20f){
          float theta = (S[q][q] - S[p][p]) / (2.f * apq);
          float tt = ((theta >= 0.f) ? 1.f : -1.f) / (fabsf(theta) + sqrtf(theta*theta + 1.f));
          float c = 1.f / sqrtf(tt*tt + 1.f);
          float sn = tt * c;
          for (int k2 = 0; k2 < 3; k2++){
            float a1 = S[k2][p], a2 = S[k2][q];
            S[k2][p] = c*a1 - sn*a2; S[k2][q] = sn*a1 + c*a2;
          }
          for (int k2 = 0; k2 < 3; k2++){
            float a1 = S[p][k2], a2 = S[q][k2];
            S[p][k2] = c*a1 - sn*a2; S[q][k2] = sn*a1 + c*a2;
          }
          for (int k2 = 0; k2 < 3; k2++){
            float a1 = V[k2][p], a2 = V[k2][q];
            V[k2][p] = c*a1 - sn*a2; V[k2][q] = sn*a1 + c*a2;
          }
        }
      }
    }
    float lam[3] = {S[0][0], S[1][1], S[2][2]};
    // sort eigenpairs descending
    for (int a1 = 0; a1 < 2; a1++)
      for (int a2 = a1 + 1; a2 < 3; a2++)
        if (lam[a2] > lam[a1]){
          float tl = lam[a1]; lam[a1] = lam[a2]; lam[a2] = tl;
          for (int k2 = 0; k2 < 3; k2++){ float tv = V[k2][a1]; V[k2][a1] = V[k2][a2]; V[k2][a2] = tv; }
        }
    float U[3][3];
    for (int j2 = 0; j2 < 3; j2++){
      float sg = sqrtf(fmaxf(lam[j2], 0.f));
      float inv = 1.f / fmaxf(sg, 1e-20f);
      for (int i2 = 0; i2 < 3; i2++)
        U[i2][j2] = (H[i2][0]*V[0][j2] + H[i2][1]*V[1][j2] + H[i2][2]*V[2][j2]) * inv;
    }
    float r0[3][3];
    for (int i2 = 0; i2 < 3; i2++)
      for (int k2 = 0; k2 < 3; k2++)
        r0[i2][k2] = V[i2][0]*U[k2][0] + V[i2][1]*U[k2][1] + V[i2][2]*U[k2][2];
    float det = r0[0][0]*(r0[1][1]*r0[2][2] - r0[1][2]*r0[2][1])
              - r0[0][1]*(r0[1][0]*r0[2][2] - r0[1][2]*r0[2][0])
              + r0[0][2]*(r0[1][0]*r0[2][1] - r0[1][1]*r0[2][0]);
    float dvec[3] = {1.f, 1.f, det};
    float R[3][3];
    for (int i2 = 0; i2 < 3; i2++)
      for (int k2 = 0; k2 < 3; k2++)
        R[i2][k2] = V[i2][0]*dvec[0]*U[k2][0] + V[i2][1]*dvec[1]*U[k2][1] + V[i2][2]*dvec[2]*U[k2][2];
    for (int i2 = 0; i2 < 3; i2++){
      for (int k2 = 0; k2 < 3; k2++) outR[b * 9 + i2 * 3 + k2] = R[i2][k2];
      outT[b * 3 + i2] = ct[i2] - (R[i2][0]*cs[0] + R[i2][1]*cs[1] + R[i2][2]*cs[2]);
    }
  }
}

// ---------------- top-256 of 1024 via bitonic sort (desc, stable ties) ----------------
__global__ __launch_bounds__(1024) void k_topk(const float* __restrict__ wgt, int* __restrict__ topi){
  __shared__ float v[1024];
  __shared__ int   id[1024];
  int b = blockIdx.x, t = threadIdx.x;
  v[t] = wgt[b * N_ + t]; id[t] = t; __syncthreads();
  for (int k = 2; k <= 1024; k <<= 1){
    for (int j = k >> 1; j > 0; j >>= 1){
      int x = t ^ j;
      if (x > t){
        bool dirDesc = ((t & k) == 0);
        bool iBefore = (v[t] > v[x]) || (v[t] == v[x] && id[t] < id[x]);
        if (dirDesc != iBefore){
          float tv = v[t]; v[t] = v[x]; v[x] = tv;
          int ti2 = id[t]; id[t] = id[x]; id[x] = ti2;
        }
      }
      __syncthreads();
    }
  }
  if (t < NKP) topi[b * NKP + t] = id[t];
}

// ---------------- gather keypoint outputs ----------------
__global__ void k_gather(const int* __restrict__ topi, const float* __restrict__ src,
                         const float* __restrict__ srcCorr, const float* __restrict__ sknnD,
                         const float* __restrict__ knnD, float* __restrict__ skp,
                         float* __restrict__ tkp, float* __restrict__ skk, float* __restrict__ tkk){
  int idx = blockIdx.x * blockDim.x + threadIdx.x;           // total B_*NKP*K_
  int kk = idx & 15;
  int j  = (idx >> 4) & (NKP - 1);
  int b  = idx >> 12;
  int ti = topi[b * NKP + j];
  #pragma unroll
  for (int c = 0; c < 3; c++){
    size_t ob = ((size_t)(b * 3 + c) * NKP + j) * K_ + kk;
    size_t ib = ((size_t)(b * N_ + ti) * K_ + kk) * 3 + c;
    skk[ob] = sknnD[ib];
    tkk[ob] = knnD[ib];
  }
  if (kk == 0){
    #pragma unroll
    for (int c = 0; c < 3; c++){
      skp[(b * 3 + c) * NKP + j] = src    [(b * 3 + c) * N_ + ti];
      tkp[(b * 3 + c) * NKP + j] = srcCorr[(b * 3 + c) * N_ + ti];
    }
  }
}

// ---------------- loss_scl ----------------
__global__ __launch_bounds__(512) void k_loss(const int* __restrict__ topi,
                                              const int* __restrict__ amax,
                                              const float* __restrict__ rmm,
                                              float* __restrict__ out){
  int t = threadIdx.x;                                       // 512 = B_*NKP
  int b = t >> 8, j = t & 255;
  int ti = topi[b * NKP + j];
  int mc = amax[b * N_ + ti];
  float p = rmm[((size_t)(b * N_ + ti)) * M_ + mc];
  float val = -logf(p + 1e-15f);
  __shared__ float sh[512];
  sh[t] = val; __syncthreads();
  for (int st = 256; st > 0; st >>= 1){ if (t < st) sh[t] += sh[t+st]; __syncthreads(); }
  if (t == 0) out[0] = sh[0] / (B_ * NKP);
}

// =====================================================================
extern "C" void kernel_launch(void* const* d_in, const int* in_sizes, int n_in,
                              void* d_out, int out_size, void* d_ws, size_t ws_size,
                              hipStream_t stream){
  (void)in_sizes; (void)n_in; (void)out_size; (void)ws_size;
  const float* src      = (const float*)d_in[0];
  const float* tgt      = (const float*)d_in[1];
  const float* semb     = (const float*)d_in[2];
  const float* temb     = (const float*)d_in[3];
  const int*   src_idx  = (const int*)  d_in[4];
  const float* src_knn  = (const float*)d_in[6];
  const int*   src_idx1 = (const int*)  d_in[9];
  const int*   idx2     = (const int*)  d_in[10];
  // disc params start at 12 (dict insertion order)
  const float* w1   = (const float*)d_in[12];
  const float* b1   = (const float*)d_in[13];
  const float* g1   = (const float*)d_in[14];
  const float* be1  = (const float*)d_in[15];
  const float* w2   = (const float*)d_in[16];
  const float* b2   = (const float*)d_in[17];
  const float* g2   = (const float*)d_in[18];
  const float* be2  = (const float*)d_in[19];
  const float* m2w1 = (const float*)d_in[20];
  const float* m2b1 = (const float*)d_in[21];
  const float* m2g1 = (const float*)d_in[22];
  const float* m2be1= (const float*)d_in[23];
  const float* m2w2 = (const float*)d_in[24];
  const float* m2b2 = (const float*)d_in[25];
  const float* m2g2 = (const float*)d_in[26];
  const float* m2be2= (const float*)d_in[27];
  const float* m3w1 = (const float*)d_in[28];
  const float* m3b1 = (const float*)d_in[29];
  const float* m3g1 = (const float*)d_in[30];
  const float* m3be1= (const float*)d_in[31];
  const float* m3w2 = (const float*)d_in[32];
  const float* m3b2 = (const float*)d_in[33];
  const float* m3g2 = (const float*)d_in[34];
  const float* m3be2= (const float*)d_in[35];
  const float* m4w1 = (const float*)d_in[36];
  const float* m4b1 = (const float*)d_in[37];
  const float* m4g1 = (const float*)d_in[38];
  const float* m4be1= (const float*)d_in[39];
  const float* m4w2 = (const float*)d_in[40];
  const float* m4b2 = (const float*)d_in[41];

  float* outF = (float*)d_out;
  float* outR  = outF;            // 18
  float* outT  = outF + 18;       // 6
  float* oSkp  = outF + 24;       // 1536
  float* oTkp  = outF + 1560;     // 1536
  float* oSkk  = outF + 3096;     // 24576
  float* oTkk  = outF + 27672;    // 24576
  float* oLoss = outF + 52248;    // 1

  // -------- workspace carve --------
  char* wp = (char*)d_ws;
  auto carve = [&](size_t bytes) -> void* {
    void* r = (void*)wp;
    wp += (bytes + 255) & ~(size_t)255;
    return r;
  };
  float* dmap    = (float*)carve((size_t)B_*N_*M_*4);
  float* scores  = (float*)carve((size_t)B_*N_*M_*4);
  float* rmm     = (float*)carve((size_t)B_*N_*M_*4);    // refined -> softmax in place
  float* xx      = (float*)carve((size_t)B_*N_*4);
  float* yy      = (float*)carve((size_t)B_*M_*4);
  u16*   srcT    = (u16*)  carve((size_t)B_*N_*C_*2);
  u16*   tgtT    = (u16*)  carve((size_t)B_*M_*C_*2);
  int*   amax    = (int*)  carve((size_t)B_*N_*4);
  float* srcCorr = (float*)carve((size_t)B_*3*N_*4);
  float* knnD    = (float*)carve((size_t)B_*N_*K_*3*4);
  float* sknnD   = (float*)carve((size_t)B_*N_*K_*3*4);
  u16*   w2b     = (u16*)  carve((size_t)D2_*3*DIM_*2);
  u16*   z1      = (u16*)  carve((size_t)B_*K_*N_*DIM_*2);
  u16*   bufA    = (u16*)  carve((size_t)B_*K_*N_*D2_*2);
  u16*   bufB    = (u16*)  carve((size_t)B_*K_*N_*D2_*2);
  float* muB     = (float*)carve(512*4);
  float* varB    = (float*)carve(512*4);
  float* c1      = (float*)carve((size_t)B_*16*N_*16*4);
  float* c2      = (float*)carve((size_t)B_*16*N_*16*4);
  float* c3      = (float*)carve((size_t)B_*16*N_*16*4);
  float* c4      = (float*)carve((size_t)B_*16*N_*16*4);
  float* sbuf    = (float*)carve((size_t)B_*16*N_*4);
  float* s2buf   = (float*)carve((size_t)B_*8*N_*4);
  float* wgt     = (float*)carve((size_t)B_*N_*4);
  int*   topi    = (int*)  carve((size_t)B_*NKP*4);

  // -------- phase 1: matching --------
  hipLaunchKernelGGL(k_embT, dim3(4096), dim3(256), 0, stream, semb, srcT);
  hipLaunchKernelGGL(k_embT, dim3(4096), dim3(256), 0, stream, temb, tgtT);
  hipLaunchKernelGGL(k_sqsum, dim3(8), dim3(256), 0, stream, semb, xx);
  hipLaunchKernelGGL(k_sqsum, dim3(8), dim3(256), 0, stream, temb, yy);
  hipLaunchKernelGGL(k_pair_wmma, dim3(32, 32, B_), dim3(32), 0, stream, srcT, tgtT, xx, yy, dmap);
  hipLaunchKernelGGL(k_softmax_neg, dim3(B_*N_), dim3(256), 0, stream, dmap, scores, (int*)nullptr);
  hipLaunchKernelGGL(k_refine, dim3(8192), dim3(256), 0, stream, scores, dmap, src_idx1, idx2, rmm);
  hipLaunchKernelGGL(k_softmax_neg, dim3(B_*N_), dim3(256), 0, stream, rmm, rmm, amax);
  hipLaunchKernelGGL(k_src_corr, dim3(B_*N_), dim3(128), 0, stream, tgt, rmm, srcCorr);
  hipLaunchKernelGGL(k_knn_dist, dim3(128), dim3(256), 0, stream, src, srcCorr, src_knn, src_idx, knnD, sknnD);

  // -------- phase 2: discriminator --------
  hipLaunchKernelGGL(k_w2conv, dim3((D2_*3*DIM_)/256), dim3(256), 0, stream, w2, w2b);
  const int rows1 = B_ * K_ * N_;
  // x path (knn_distance)
  hipLaunchKernelGGL(k_conv1, dim3(rows1), dim3(256), 0, stream, knnD, w1, b1, z1);
  hipLaunchKernelGGL(k_stats_last, dim3(DIM_), dim3(256), 0, stream, z1, DIM_, rows1, muB, varB);
  hipLaunchKernelGGL(k_bn_lrelu_last, dim3((rows1*DIM_)/256), dim3(256), 0, stream, z1, g1, be1, muB, varB, DIM_);
  hipLaunchKernelGGL(k_conv2_wmma, dim3(32, 16, B_*K_), dim3(32), 0, stream, z1, w2b, b2, bufA);
  hipLaunchKernelGGL(k_stats_last, dim3(D2_), dim3(256), 0, stream, bufA, D2_, rows1, muB, varB);
  hipLaunchKernelGGL(k_bn_lrelu_last, dim3((rows1*D2_)/256), dim3(256), 0, stream, bufA, g2, be2, muB, varB, D2_);
  // y path (src_knn_distance), fused subtract into bufA => x2
  hipLaunchKernelGGL(k_conv1, dim3(rows1), dim3(256), 0, stream, sknnD, w1, b1, z1);
  hipLaunchKernelGGL(k_stats_last, dim3(DIM_), dim3(256), 0, stream, z1, DIM_, rows1, muB, varB);
  hipLaunchKernelGGL(k_bn_lrelu_last, dim3((rows1*DIM_)/256), dim3(256), 0, stream, z1, g1, be1, muB, varB, DIM_);
  hipLaunchKernelGGL(k_conv2_wmma, dim3(32, 16, B_*K_), dim3(32), 0, stream, z1, w2b, b2, bufB);
  hipLaunchKernelGGL(k_stats_last, dim3(D2_), dim3(256), 0, stream, bufB, D2_, rows1, muB, varB);
  hipLaunchKernelGGL(k_bn_lrelu_sub, dim3((rows1*D2_)/256), dim3(256), 0, stream, bufA, bufB, g2, be2, muB, varB);

  // m2 / m3 heads
  const int nOut = B_ * 16 * N_ * 16;
  const int HW = N_ * 16;
  hipLaunchKernelGGL(k_conv1x1_bf, dim3(nOut/256), dim3(256), 0, stream, bufA, m2w1, m2b1, c1);
  hipLaunchKernelGGL(k_stats_chw, dim3(16), dim3(256), 0, stream, c1, 16, HW, muB, varB);
  hipLaunchKernelGGL(k_bn_lrelu_chw, dim3(nOut/256), dim3(256), 0, stream, c1, m2g1, m2be1, muB, varB, 16, HW);
  hipLaunchKernelGGL(k_conv1x1_f32, dim3(nOut/256), dim3(256), 0, stream, c1, m2w2, m2b2, c2);
  hipLaunchKernelGGL(k_stats_chw, dim3(16), dim3(256), 0, stream, c2, 16, HW, muB, varB);
  hipLaunchKernelGGL(k_bn_lrelu_chw, dim3(nOut/256), dim3(256), 0, stream, c2, m2g2, m2be2, muB, varB, 16, HW);
  hipLaunchKernelGGL(k_conv1x1_bf, dim3(nOut/256), dim3(256), 0, stream, bufA, m3w1, m3b1, c3);
  hipLaunchKernelGGL(k_stats_chw, dim3(16), dim3(256), 0, stream, c3, 16, HW, muB, varB);
  hipLaunchKernelGGL(k_bn_lrelu_chw, dim3(nOut/256), dim3(256), 0, stream, c3, m3g1, m3be1, muB, varB, 16, HW);
  hipLaunchKernelGGL(k_conv1x1_f32, dim3(nOut/256), dim3(256), 0, stream, c3, m3w2, m3b2, c4);
  hipLaunchKernelGGL(k_stats_chw, dim3(16), dim3(256), 0, stream, c4, 16, HW, muB, varB);
  hipLaunchKernelGGL(k_bn_lrelu_chw, dim3(nOut/256), dim3(256), 0, stream, c4, m3g2, m3be2, muB, varB, 16, HW);
  hipLaunchKernelGGL(k_softmax16, dim3((B_*16*N_)/256), dim3(256), 0, stream, c4);
  hipLaunchKernelGGL(k_wsum, dim3((B_*16*N_)/256), dim3(256), 0, stream, c2, c4, sbuf);
  hipLaunchKernelGGL(k_conv1d, dim3((B_*8*N_)/256), dim3(256), 0, stream, sbuf, m4w1, m4b1, s2buf, 16, 8);
  hipLaunchKernelGGL(k_stats_chw, dim3(8), dim3(256), 0, stream, s2buf, 8, N_, muB, varB);
  hipLaunchKernelGGL(k_bn_lrelu_chw, dim3((B_*8*N_)/256), dim3(256), 0, stream, s2buf, m4g1, m4be1, muB, varB, 8, N_);
  hipLaunchKernelGGL(k_final_weight, dim3((B_*N_)/256), dim3(256), 0, stream, s2buf, m4w2, m4b2, wgt);

  // -------- phase 3: rigid transform, top-k, outputs --------
  hipLaunchKernelGGL(k_rigid, dim3(B_), dim3(256), 0, stream, src, srcCorr, wgt, outR, outT);
  hipLaunchKernelGGL(k_topk, dim3(B_), dim3(1024), 0, stream, wgt, topi);
  hipLaunchKernelGGL(k_gather, dim3((B_*NKP*K_)/256), dim3(256), 0, stream,
                     topi, src, srcCorr, sknnD, knnD, oSkp, oTkp, oSkk, oTkk);
  hipLaunchKernelGGL(k_loss, dim3(1), dim3(512), 0, stream, topi, amax, rmm, oLoss);
}
